// SparseTransformerLayer_77214922047595
// MI455X (gfx1250) — compile-verified, matrix operand order checked by reading, not run on hardware
//
#include <hip/hip_runtime.h>
#include <hip/hip_bf16.h>
#include <math.h>

typedef __attribute__((ext_vector_type(16))) _Float16 v16h;
typedef __attribute__((ext_vector_type(8)))  _Float16 v8h;
typedef __attribute__((ext_vector_type(8)))  float    v8f;
typedef int v4i_b __attribute__((vector_size(16)));   // matches builtin param pointee

#define C_DIM 256
#define KNN   32
#define HEADS 8

#ifndef __has_builtin
#define __has_builtin(x) 0
#endif
#if __has_builtin(__builtin_amdgcn_global_load_async_to_lds_b128)
#define ASYNC_LDS 1
#endif

// ---------------------------------------------------------------------------
// Elementwise f32 -> f16 conversion (4 elems/thread, n multiple of 1024)
// ---------------------------------------------------------------------------
__global__ __launch_bounds__(256)
void cvt_f32_to_f16_kernel(const float* __restrict__ in, _Float16* __restrict__ out, int n)
{
    const int i = (blockIdx.x * 256 + threadIdx.x) * 4;
    if (i >= n) return;
    const float4 v = *(const float4*)(in + i);
    out[i + 0] = (_Float16)v.x;
    out[i + 1] = (_Float16)v.y;
    out[i + 2] = (_Float16)v.z;
    out[i + 3] = (_Float16)v.w;
}

// ---------------------------------------------------------------------------
// gfx1250 16-bit WMMA fragment: lane&15 selects the row (A: M, B: N);
// lanes 0-15 hold K = {k0..k0+7, k0+16..k0+23}, lanes 16-31 hold
// K = {k0+8..k0+15, k0+24..k0+31}.  Two aligned 16B half8 loads.
// Works for both global (global_load_b128) and LDS (ds_load_b128) pointers.
// ---------------------------------------------------------------------------
__device__ __forceinline__ v16h ldfrag(const _Float16* __restrict__ row, int c0, int c1)
{
    const v8h lo = *(const v8h*)(row + c0);
    const v8h hi = *(const v8h*)(row + c1);
    return __builtin_shufflevector(lo, hi, 0,1,2,3,4,5,6,7,8,9,10,11,12,13,14,15);
}

__device__ __forceinline__ v8f wmma_step(v16h a, v16h b, v8f c)
{
    return __builtin_amdgcn_wmma_f32_16x16x32_f16(false, a, false, b, (short)0, c, false, false);
}

// ---------------------------------------------------------------------------
// GEMM: Y(rows,Cout) = Xh(rows,CIN) @ Wh(Cout,CIN)^T + bias  [optional ReLU]
// Block = 256 threads (8 waves) -> 256M x 64N block tile.
// Weight K-tile (64 x 256 f16, padded rows) is staged in LDS once per block
// via async global->LDS DMA (ASYNCcnt) and reused by all waves; each wave
// owns a 32M x 64N strip -> 8 v_wmma per K-step against 2 streaming global
// A fragments + 4 LDS B fragments.
// ---------------------------------------------------------------------------
template<int CIN>
__global__ __launch_bounds__(256)
void gemm_wmma_f16(const _Float16* __restrict__ X, const _Float16* __restrict__ W,
                   const float* __restrict__ bias,
                   float* __restrict__ Yf, _Float16* __restrict__ Yh,
                   int Cout, int relu)
{
    constexpr int KT   = 256;          // K tile staged in LDS
    constexpr int LSTR = KT + 8;       // padded row stride (halves): 528B rows
    __shared__ _Float16 ldsB[64 * LSTR];   // ~33.8 KB

    const int tid  = threadIdx.x;
    const int wave = tid >> 5;
    const int lane = tid & 31;
    const int r    = lane & 15;
    const int grp  = lane >> 4;
    const int tM   = blockIdx.x * 256 + wave * 32;
    const int tN   = blockIdx.y * 64;

    const _Float16* xrow0 = X + (size_t)(tM +      r) * CIN;
    const _Float16* xrow1 = X + (size_t)(tM + 16 + r) * CIN;
    __builtin_prefetch(xrow0, 0, 3);   // global_prefetch_b8
    __builtin_prefetch(xrow1, 0, 3);

    v8f acc[2][4] = {};

#pragma unroll
    for (int kb = 0; kb < CIN; kb += KT) {
        // ---- cooperative stage of the 64 x KT weight tile into LDS
#pragma unroll
        for (int i = 0; i < 8; ++i) {
            const int chunk = tid + i * 256;          // 2048 chunks of 8 halves
            const int row   = chunk >> 5;
            const int col   = (chunk & 31) << 3;
            const _Float16* g = W + (size_t)(tN + row) * CIN + kb + col;
            _Float16*       l = &ldsB[row * LSTR + col];
#if defined(ASYNC_LDS)
            __builtin_amdgcn_global_load_async_to_lds_b128(
                (__attribute__((address_space(1))) v4i_b*)g,
                (__attribute__((address_space(3))) v4i_b*)l, 0, 0);
#else
            *(v8h*)l = *(const v8h*)g;
#endif
        }
#if defined(ASYNC_LDS)
#if __has_builtin(__builtin_amdgcn_s_wait_asynccnt)
        __builtin_amdgcn_s_wait_asynccnt(0);
#else
        asm volatile("s_wait_asynccnt 0" ::: "memory");
#endif
#endif
        __syncthreads();

        // ---- 8 K-steps of 32: ds B fragments, streaming global A fragments
#pragma unroll
        for (int kk = 0; kk < KT; kk += 32) {
            const int c0 = kk + grp * 8;
            const int c1 = kk + 16 + grp * 8;
            const v16h a0 = ldfrag(xrow0 + kb, c0, c1);
            const v16h a1 = ldfrag(xrow1 + kb, c0, c1);
#pragma unroll
            for (int t = 0; t < 4; ++t) {
                const v16h b = ldfrag(&ldsB[(16 * t + r) * LSTR], c0, c1);
                acc[0][t] = wmma_step(a0, b, acc[0][t]);
                acc[1][t] = wmma_step(a1, b, acc[1][t]);
            }
        }
        __syncthreads();
    }

    // ---- epilogue: C/D layout VGPR rr -> M = base + rr (+8 for lanes>=16)
#pragma unroll
    for (int t = 0; t < 4; ++t) {
        const int   ncol = tN + 16 * t + r;
        const float bv   = bias[ncol];
#pragma unroll
        for (int s = 0; s < 2; ++s) {
#pragma unroll
            for (int rr = 0; rr < 8; ++rr) {
                const int m = tM + 16 * s + rr + (grp << 3);
                float v = acc[s][t][rr] + bv;
                if (relu) v = fmaxf(v, 0.0f);
                const size_t o = (size_t)m * Cout + ncol;
                if (Yf) Yf[o] = v;
                if (Yh) Yh[o] = (_Float16)v;
            }
        }
    }
}

// ---------------------------------------------------------------------------
// Sparse neighborhood attention: one wave32 per query.
// ---------------------------------------------------------------------------
__global__ __launch_bounds__(256)
void attn_kernel(const float* __restrict__ q,   const float* __restrict__ kp,
                 const float* __restrict__ vp,  const int*   __restrict__ indices,
                 const float* __restrict__ icoord, const float* __restrict__ mcoord,
                 const float* __restrict__ mask,
                 const float* __restrict__ Wrpe, const float* __restrict__ brpe,
                 _Float16* __restrict__ hidden, int N)
{
    __shared__ float attnLDS[8][KNN * HEADS];

    const int wave = threadIdx.x >> 5;
    const int lane = threadIdx.x & 31;
    const int n    = blockIdx.x * 8 + wave;
    if (n >= N) return;

    const int   idx  = indices[(size_t)n * KNN + lane];
    const float rx   = mcoord[idx * 3 + 0] - icoord[n * 3 + 0];
    const float ry   = mcoord[idx * 3 + 1] - icoord[n * 3 + 1];
    const float rz   = mcoord[idx * 3 + 2] - icoord[n * 3 + 2];
    const float* krow = kp + (size_t)idx * C_DIM;
    const float* qrow = q  + (size_t)n   * C_DIM;

    const float scale = 0.17677669529663687f;   // 1/sqrt(32)
    const float mpen  = 1.0e6f * (1.0f - mask[(size_t)n * KNN + lane]);

#pragma unroll
    for (int h = 0; h < HEADS; ++h) {
        float s = 0.0f;
#pragma unroll
        for (int pp = 0; pp < 16; ++pp) {
            const int p = h * 16 + pp;
            const float ang = Wrpe[p * 3 + 0] * rx + Wrpe[p * 3 + 1] * ry +
                              Wrpe[p * 3 + 2] * rz + brpe[p];
            const float cs = __cosf(ang), sn = __sinf(ang);
            const float x0 = krow[2 * p], x1 = krow[2 * p + 1];
            const float k0 = x0 * cs - x1 * sn;
            const float k1 = x0 * sn + x1 * cs;
            s += qrow[2 * p] * k0 + qrow[2 * p + 1] * k1;
        }
        s = s * scale - mpen;
        float m = s;
#pragma unroll
        for (int off = 16; off; off >>= 1) m = fmaxf(m, __shfl_xor(m, off, 32));
        const float e = __expf(s - m);
        float sum = e;
#pragma unroll
        for (int off = 16; off; off >>= 1) sum += __shfl_xor(sum, off, 32);
        attnLDS[wave][lane * HEADS + h] = e / sum;
    }

    const int h = lane >> 2;
    float a0 = 0.f, a1 = 0.f, a2 = 0.f, a3 = 0.f;
    float a4 = 0.f, a5 = 0.f, a6 = 0.f, a7 = 0.f;
    for (int j = 0; j < KNN; ++j) {
        const int   ij = __shfl(idx, j, 32);
        const float w  = attnLDS[wave][j * HEADS + h];
        const float4* vr = (const float4*)(vp + (size_t)ij * C_DIM + lane * 8);
        const float4 v0 = vr[0], v1 = vr[1];
        a0 += w * v0.x; a1 += w * v0.y; a2 += w * v0.z; a3 += w * v0.w;
        a4 += w * v1.x; a5 += w * v1.y; a6 += w * v1.z; a7 += w * v1.w;
    }
    v8h hv;
    hv[0]=(_Float16)a0; hv[1]=(_Float16)a1; hv[2]=(_Float16)a2; hv[3]=(_Float16)a3;
    hv[4]=(_Float16)a4; hv[5]=(_Float16)a5; hv[6]=(_Float16)a6; hv[7]=(_Float16)a7;
    *(v8h*)(hidden + (size_t)n * C_DIM + lane * 8) = hv;
}

// ---------------------------------------------------------------------------
// Residual + LayerNorm over C=256: one 256-thread block per row.
// ---------------------------------------------------------------------------
__global__ __launch_bounds__(256)
void ln_residual_kernel(const float* __restrict__ x, const float* __restrict__ res,
                        const float* __restrict__ g, const float* __restrict__ b,
                        float* __restrict__ out, _Float16* __restrict__ out16)
{
    __shared__ float red[256];
    const int row = blockIdx.x, t = threadIdx.x;
    const size_t base = (size_t)row * C_DIM + t;
    const float v = x[base] + res[base];

    red[t] = v; __syncthreads();
#pragma unroll
    for (int off = 128; off; off >>= 1) { if (t < off) red[t] += red[t + off]; __syncthreads(); }
    const float mu = red[0] * (1.0f / C_DIM);
    __syncthreads();

    const float d = v - mu;
    red[t] = d * d; __syncthreads();
#pragma unroll
    for (int off = 128; off; off >>= 1) { if (t < off) red[t] += red[t + off]; __syncthreads(); }
    const float var = red[0] * (1.0f / C_DIM);

    const float y = d * rsqrtf(var + 1e-5f) * g[t] + b[t];
    out[base] = y;
    if (out16) out16[base] = (_Float16)y;
}

// ---------------------------------------------------------------------------
// Launcher
// ---------------------------------------------------------------------------
extern "C" void kernel_launch(void* const* d_in, const int* in_sizes, int n_in,
                              void* d_out, int out_size, void* d_ws, size_t ws_size,
                              hipStream_t stream)
{
    const float* input_states = (const float*)d_in[0];
    const float* memory_states= (const float*)d_in[1];
    const int*   indices      = (const int*)  d_in[2];
    const float* input_coord  = (const float*)d_in[3];
    const float* memory_coord = (const float*)d_in[4];
    const float* attn_mask    = (const float*)d_in[5];
    const float* Wq = (const float*)d_in[6];  const float* bq = (const float*)d_in[7];
    const float* Wk = (const float*)d_in[8];  const float* bk = (const float*)d_in[9];
    const float* Wv = (const float*)d_in[10]; const float* bv = (const float*)d_in[11];
    const float* Wo = (const float*)d_in[12]; const float* bo = (const float*)d_in[13];
    const float* ln1_g = (const float*)d_in[14]; const float* ln1_b = (const float*)d_in[15];
    const float* Wrpe  = (const float*)d_in[16]; const float* brpe  = (const float*)d_in[17];
    const float* We = (const float*)d_in[18]; const float* be = (const float*)d_in[19];
    const float* Ws = (const float*)d_in[20]; const float* bs = (const float*)d_in[21];
    const float* ln2_g = (const float*)d_in[22]; const float* ln2_b = (const float*)d_in[23];

    const int N = 8192, C = C_DIM;
    const size_t T = (size_t)N * C;              // 2M elements

    float*    ws    = (float*)d_ws;
    float*    qb    = ws;                        // q  -> proj -> ff2
    float*    kpb   = ws + T;                    // kproj -> ff1(f16)
    float*    vpb   = ws + 2 * T;                // vproj -> out16(f16)
    _Float16* hid_h = (_Float16*)(ws + 3 * T);   // T halves
    float*    outb  = ws + 3 * T + T / 2;        // post-LN1 f32
    _Float16* Xh    = (_Float16*)(ws + 4 * T + T / 2);
    _Float16* Mh    = (_Float16*)(ws + 5 * T);
    _Float16* wpool = (_Float16*)(ws + 5 * T + T / 2);
    _Float16* Wqh = wpool;            _Float16* Wkh = wpool + 65536;
    _Float16* Wvh = wpool + 131072;   _Float16* Woh = wpool + 196608;
    _Float16* Weh = wpool + 262144;   _Float16* Wsh = wpool + 393216;

    _Float16* ff1_h = (_Float16*)kpb;
    _Float16* out16 = (_Float16*)vpb;
    float*    ff2   = qb;

    dim3 blk(256);

    // ---- f32 -> f16 staging
    cvt_f32_to_f16_kernel<<<(int)(T / 1024), blk, 0, stream>>>(input_states,  Xh,  (int)T);
    cvt_f32_to_f16_kernel<<<(int)(T / 1024), blk, 0, stream>>>(memory_states, Mh,  (int)T);
    cvt_f32_to_f16_kernel<<<64,  blk, 0, stream>>>(Wq, Wqh, 65536);
    cvt_f32_to_f16_kernel<<<64,  blk, 0, stream>>>(Wk, Wkh, 65536);
    cvt_f32_to_f16_kernel<<<64,  blk, 0, stream>>>(Wv, Wvh, 65536);
    cvt_f32_to_f16_kernel<<<64,  blk, 0, stream>>>(Wo, Woh, 65536);
    cvt_f32_to_f16_kernel<<<128, blk, 0, stream>>>(We, Weh, 131072);
    cvt_f32_to_f16_kernel<<<128, blk, 0, stream>>>(Ws, Wsh, 131072);

    // ---- Q/K/V projections (WMMA, LDS-staged weights)
    gemm_wmma_f16<256><<<dim3(32, 4), blk, 0, stream>>>(Xh, Wqh, bq, qb,  nullptr, C, 0);
    gemm_wmma_f16<256><<<dim3(32, 4), blk, 0, stream>>>(Mh, Wkh, bk, kpb, nullptr, C, 0);
    gemm_wmma_f16<256><<<dim3(32, 4), blk, 0, stream>>>(Mh, Wvh, bv, vpb, nullptr, C, 0);

    // ---- gather + rotary + neighborhood softmax + attn*V
    attn_kernel<<<N / 8, blk, 0, stream>>>(qb, kpb, vpb, indices, input_coord,
                                           memory_coord, attn_mask, Wrpe, brpe, hid_h, N);

    // ---- output projection + residual LN1
    gemm_wmma_f16<256><<<dim3(32, 4), blk, 0, stream>>>(hid_h, Woh, bo, qb, nullptr, C, 0);
    ln_residual_kernel<<<N, blk, 0, stream>>>(qb, input_states, ln1_g, ln1_b, outb, out16);

    // ---- FFN
    gemm_wmma_f16<256><<<dim3(32, 8), blk, 0, stream>>>(out16, Weh, be, nullptr, ff1_h, 2 * C, 1);
    gemm_wmma_f16<512><<<dim3(32, 4), blk, 0, stream>>>(ff1_h, Wsh, bs, ff2, nullptr, C, 0);
    ln_residual_kernel<<<N, blk, 0, stream>>>(ff2, outb, ln2_g, ln2_b, (float*)d_out, nullptr);
}
